// Attention_72730976190889
// MI455X (gfx1250) — matrix operand order repaired and checked by reading, not compiled, not run on hardware
//
#include <hip/hip_runtime.h>

// ---------------------------------------------------------------------------
// Transformer-XL relative-position attention for MI455X (gfx1250), wave32.
// All matmuls run on v_wmma_f32_16x16x32_f16 (f16 in, f32 accumulate).
// The f16->f32 attention GEMMs (scores, rel-logits) use double-buffered
// GLOBAL_LOAD_ASYNC_TO_LDS_B128 staging tracked with ASYNCcnt.
// ---------------------------------------------------------------------------

typedef _Float16 half_t;
typedef __attribute__((ext_vector_type(16))) _Float16 v16h;
typedef __attribute__((ext_vector_type(8)))  _Float16 v8h;
typedef __attribute__((ext_vector_type(8)))  float    v8f;

constexpr int kB  = 2;
constexpr int kT  = 2048;
constexpr int kD  = 1024;
constexpr int kH  = 16;
constexpr int kHD = 64;

constexpr int BM   = 64;    // block tile M
constexpr int BN   = 128;   // block tile N
constexpr int BK   = 32;    // k-step (matches 16x16x32 WMMA)
constexpr int BKP  = 40;    // padded LDS row stride in halves (80B = 5*16B)
constexpr int NTHR = 256;   // 8 waves of 32

template <class X, class Y> struct is_same_t { static constexpr bool v = false; };
template <class X> struct is_same_t<X, X>    { static constexpr bool v = true;  };

// ---- WMMA fragment loaders from LDS (documented gfx1250 layouts) ----------
// A 16x32 f16: lane r=lane&15 is row M=r; halves 0..7 hold K = hi*8 + 0..7,
// halves 8..15 hold K = 16 + hi*8 + 0..7  (hi = lane>>4).
static __device__ __forceinline__ v16h frag_a_ld(const half_t* base) {
  const int lane = threadIdx.x & 31;
  const int r    = lane & 15;
  const int hi   = lane >> 4;
  const half_t* p = base + r * BKP + hi * 8;
  union { v16h v; v8h h[2]; } u;
  u.h[0] = *(const v8h*)(p);
  u.h[1] = *(const v8h*)(p + 16);
  return u.v;
}
// B 32x16 f16: lane r=lane&15 is column N=r; halves 0..7 hold K = hi*8 + 0..7,
// halves 8..15 hold K = 16 + hi*8 + 0..7  (hi = lane>>4).
static __device__ __forceinline__ v16h frag_b_ld(const half_t* base) {
  const int lane = threadIdx.x & 31;
  const int r    = lane & 15;
  const int hi   = lane >> 4;
  const half_t* p = base + r * BKP + hi * 8;
  union { v16h v; v8h h[2]; } u;
  u.h[0] = *(const v8h*)(p);
  u.h[1] = *(const v8h*)(p + 16);
  return u.v;
}

// ---- Async global->LDS 16B copy (CDNA5 TDM-lite path, ASYNCcnt-tracked) ---
// Flat LDS-aperture addresses carry the LDS byte offset in the low 32 bits
// (ISA 10.2 aperture table), which is exactly what the async-load VDST wants:
// an LDS byte offset held in a VGPR.
static __device__ __forceinline__ void async_cp16(void* lds, const void* g) {
  const unsigned loff = (unsigned)(unsigned long long)(uintptr_t)lds;
  const unsigned long long ga = (unsigned long long)(uintptr_t)g;
  asm volatile("global_load_async_to_lds_b128 %0, %1, off"
               :: "v"(loff), "v"(ga) : "memory");
}
static __device__ __forceinline__ void async_wait_all() {
  asm volatile("s_wait_asynccnt 0x0" ::: "memory");
}
static __device__ __forceinline__ void async_wait_keep3() {
  asm volatile("s_wait_asynccnt 0x3" ::: "memory");
}

// Stage a BMxBK A-tile and BNxBK B-tile (both f16, k-contiguous rows) into
// LDS with 3 async instructions per wave (1 A chunk + 2 B chunks per thread).
// Requires M,N,K exact multiples of the tile sizes (true for all callers).
static __device__ __forceinline__ void stage_async_bt(
    half_t (*As)[BKP], half_t (*Bs)[BKP],
    const half_t* __restrict__ A, int lda,
    const half_t* __restrict__ B, int ldb,
    int m0, int n0, int kb)
{
  const int tid = threadIdx.x;
  {
    const int tr = tid >> 2;           // 0..63
    const int kc = (tid & 3) * 8;      // 0,8,16,24
    async_cp16(&As[tr][kc], A + (size_t)(m0 + tr) * lda + (kb + kc));
  }
  {
    const int nr = tid >> 1;           // 0..127
    const int kc = (tid & 1) * 16;     // 0,16
    const half_t* p = B + (size_t)(n0 + nr) * ldb + (kb + kc);
    async_cp16(&Bs[nr][kc],     p);
    async_cp16(&Bs[nr][kc + 8], p + 8);
  }
}

// ---------------------------------------------------------------------------
// Generic tiled WMMA GEMM:  C[M,N] = A[M,K] * B + bias
//   BTRANS=true : B is [N,K] row-major (k-contiguous)  -> computes A*B^T
//   BTRANS=false: B is [K,N] row-major (n-contiguous)
// A/B converted to f16 while staging into LDS; accumulate f32; store OT.
// ---------------------------------------------------------------------------
template <typename AT, typename BT, typename OT, bool BTRANS>
__global__ __launch_bounds__(NTHR) void gemm_wmma(
    const AT* __restrict__ A, int lda, const BT* __restrict__ B, int ldb,
    OT* __restrict__ C, int ldc, const float* __restrict__ bias,
    int M, int N, int K)
{
  __shared__ __align__(16) half_t As[BM][BKP];
  __shared__ __align__(16) half_t Bs[BN][BKP];

  const int m0   = blockIdx.x * BM;
  const int n0   = blockIdx.y * BN;
  const int tid  = threadIdx.x;
  const int lane = tid & 31;
  const int wave = tid >> 5;
  const int wm   = wave & 3;   // 4 waves along M
  const int wn   = wave >> 2;  // 2 waves along N

  v8f zero = {};
  v8f acc[4] = {zero, zero, zero, zero};

  for (int kb = 0; kb < K; kb += BK) {
    __syncthreads();
    // ---- stage A tile (BM x BK), convert to f16 ----
    {
      const int tr   = tid >> 2;           // 0..63
      const int kc   = (tid & 3) * 8;      // 0,8,16,24
      const int grow = m0 + tr;
      v8h av = {};
      if (grow < M) {
        const AT* p = A + (size_t)grow * lda + (kb + kc);
        if constexpr (is_same_t<AT, float>::v) {
          float4 f0 = *(const float4*)(p);
          float4 f1 = *(const float4*)(p + 4);
          av[0]=(half_t)f0.x; av[1]=(half_t)f0.y; av[2]=(half_t)f0.z; av[3]=(half_t)f0.w;
          av[4]=(half_t)f1.x; av[5]=(half_t)f1.y; av[6]=(half_t)f1.z; av[7]=(half_t)f1.w;
        } else {
          av = *(const v8h*)(p);
        }
      }
      *(v8h*)(&As[tr][kc]) = av;
    }
    // ---- stage B tile (stored as Bs[n][k]) ----
    if constexpr (BTRANS) {
      const int nr = tid >> 1;             // 0..127
      const int kc = (tid & 1) * 16;       // 0,16
      const int gn = n0 + nr;
      v8h b0 = {}, b1 = {};
      if (gn < N) {
        const BT* p = B + (size_t)gn * ldb + (kb + kc);
        if constexpr (is_same_t<BT, float>::v) {
          float4 f;
          f = *(const float4*)(p + 0);  b0[0]=(half_t)f.x; b0[1]=(half_t)f.y; b0[2]=(half_t)f.z; b0[3]=(half_t)f.w;
          f = *(const float4*)(p + 4);  b0[4]=(half_t)f.x; b0[5]=(half_t)f.y; b0[6]=(half_t)f.z; b0[7]=(half_t)f.w;
          f = *(const float4*)(p + 8);  b1[0]=(half_t)f.x; b1[1]=(half_t)f.y; b1[2]=(half_t)f.z; b1[3]=(half_t)f.w;
          f = *(const float4*)(p + 12); b1[4]=(half_t)f.x; b1[5]=(half_t)f.y; b1[6]=(half_t)f.z; b1[7]=(half_t)f.w;
        } else {
          b0 = *(const v8h*)(p);
          b1 = *(const v8h*)(p + 8);
        }
      }
      *(v8h*)(&Bs[nr][kc])     = b0;
      *(v8h*)(&Bs[nr][kc + 8]) = b1;
    } else {
      const int kr = tid >> 3;             // 0..31
      const int nc = (tid & 7) * 16;       // 0..112
      const int gk = kb + kr;
      const BT* p  = B + (size_t)gk * ldb + (n0 + nc);
      if (n0 + nc + 16 <= N) {             // fast vector path, transpose via LDS
        if constexpr (is_same_t<BT, float>::v) {
          #pragma unroll
          for (int q = 0; q < 4; ++q) {
            float4 f = *(const float4*)(p + q * 4);
            Bs[nc + q*4 + 0][kr] = (half_t)f.x;
            Bs[nc + q*4 + 1][kr] = (half_t)f.y;
            Bs[nc + q*4 + 2][kr] = (half_t)f.z;
            Bs[nc + q*4 + 3][kr] = (half_t)f.w;
          }
        } else {
          v8h h0 = *(const v8h*)(p);
          v8h h1 = *(const v8h*)(p + 8);
          #pragma unroll
          for (int q = 0; q < 8; ++q) {
            Bs[nc + q][kr]     = h0[q];
            Bs[nc + 8 + q][kr] = h1[q];
          }
        }
      } else {                              // guarded tail path
        #pragma unroll
        for (int q = 0; q < 16; ++q) {
          half_t v = (half_t)0.0f;
          if (n0 + nc + q < N) v = (half_t)p[q];
          Bs[nc + q][kr] = v;
        }
      }
    }
    __syncthreads();
    // ---- compute: each wave does 16x64 via 4 WMMAs sharing one A frag ----
    v16h a = frag_a_ld(&As[wm * 16][0]);
    #pragma unroll
    for (int j = 0; j < 4; ++j) {
      v16h b = frag_b_ld(&Bs[wn * 64 + j * 16][0]);
      acc[j] = __builtin_amdgcn_wmma_f32_16x16x32_f16(
          false, a, false, b, (short)0, acc[j], false, false);
    }
  }

  // ---- epilogue: C layout is (m = i + 8*(lane>=16), n = lane&15) ----
  const int nn = lane & 15;
  const int mh = lane >> 4;
  #pragma unroll
  for (int j = 0; j < 4; ++j) {
    const int col = n0 + wn * 64 + j * 16 + nn;
    if (col >= N) continue;
    const float bv = bias ? bias[col] : 0.0f;
    #pragma unroll
    for (int i = 0; i < 8; ++i) {
      const int row = m0 + wm * 16 + mh * 8 + i;
      if (row < M) C[(size_t)row * ldc + col] = (OT)(acc[j][i] + bv);
    }
  }
}

// ---------------------------------------------------------------------------
// Async double-buffered f16 GEMM, A*B^T form (both operands k-contiguous).
// Used for R = (q + r_r_bias) * PE^T.  M,N multiples of BM/BN, K of BK.
// ---------------------------------------------------------------------------
template <typename OT>
__global__ __launch_bounds__(NTHR) void gemm_bt16_async(
    const half_t* __restrict__ A, int lda, const half_t* __restrict__ B, int ldb,
    OT* __restrict__ C, int ldc, int M, int N, int K)
{
  __shared__ __align__(16) half_t As[2][BM][BKP];
  __shared__ __align__(16) half_t Bs[2][BN][BKP];

  const int m0   = blockIdx.x * BM;
  const int n0   = blockIdx.y * BN;
  const int lane = threadIdx.x & 31;
  const int wave = threadIdx.x >> 5;
  const int wm   = wave & 3;
  const int wn   = wave >> 2;

  v8f zero = {};
  v8f acc[4] = {zero, zero, zero, zero};

  const int nit = K / BK;
  stage_async_bt(As[0], Bs[0], A, lda, B, ldb, m0, n0, 0);
  for (int it = 0; it < nit; ++it) {
    const int cur = it & 1;
    const bool has_next = (it + 1 < nit);
    if (has_next) {
      stage_async_bt(As[cur ^ 1], Bs[cur ^ 1], A, lda, B, ldb, m0, n0, (it + 1) * BK);
      async_wait_keep3();   // current buffer landed; next 3 copies stay in flight
    } else {
      async_wait_all();
    }
    __syncthreads();
    v16h a = frag_a_ld(&As[cur][wm * 16][0]);
    #pragma unroll
    for (int j = 0; j < 4; ++j) {
      v16h b = frag_b_ld(&Bs[cur][wn * 64 + j * 16][0]);
      acc[j] = __builtin_amdgcn_wmma_f32_16x16x32_f16(
          false, a, false, b, (short)0, acc[j], false, false);
    }
    __syncthreads();        // everyone done reading 'cur' before it is restaged
  }

  const int nn = lane & 15;
  const int mh = lane >> 4;
  #pragma unroll
  for (int j = 0; j < 4; ++j) {
    const int col = n0 + wn * 64 + j * 16 + nn;
    #pragma unroll
    for (int i = 0; i < 8; ++i) {
      const int row = m0 + wm * 16 + mh * 8 + i;
      C[(size_t)row * ldc + col] = (OT)(acc[j][i]);
    }
  }
}

// ---------------------------------------------------------------------------
// Score kernel: S = ((Qw * K^T) + relative_shift(R)) * scale, masked.
// Async double-buffered staging. Exact _relative_shift index map (mask is
// not causal, so the wrapped rows are kept):
//   w <= u   : R[u,   T-1-u+w]
//   w == u+1 : 0
//   w >  u+1 : R[u+1, w-u-2]
// ---------------------------------------------------------------------------
__global__ __launch_bounds__(NTHR) void score_kernel(
    const half_t* __restrict__ Aq, int lda, const half_t* __restrict__ Bk, int ldb,
    const float* __restrict__ R, const float* __restrict__ maskp,
    float* __restrict__ S, int T, int K, float scale)
{
  __shared__ __align__(16) half_t As[2][BM][BKP];
  __shared__ __align__(16) half_t Bs[2][BN][BKP];

  const int m0   = blockIdx.x * BM;
  const int n0   = blockIdx.y * BN;
  const int lane = threadIdx.x & 31;
  const int wave = threadIdx.x >> 5;
  const int wm   = wave & 3;
  const int wn   = wave >> 2;

  v8f zero = {};
  v8f acc[4] = {zero, zero, zero, zero};

  const int nit = K / BK;
  stage_async_bt(As[0], Bs[0], Aq, lda, Bk, ldb, m0, n0, 0);
  for (int it = 0; it < nit; ++it) {
    const int cur = it & 1;
    const bool has_next = (it + 1 < nit);
    if (has_next) {
      stage_async_bt(As[cur ^ 1], Bs[cur ^ 1], Aq, lda, Bk, ldb, m0, n0, (it + 1) * BK);
      async_wait_keep3();
    } else {
      async_wait_all();
    }
    __syncthreads();
    v16h a = frag_a_ld(&As[cur][wm * 16][0]);
    #pragma unroll
    for (int j = 0; j < 4; ++j) {
      v16h b = frag_b_ld(&Bs[cur][wn * 64 + j * 16][0]);
      acc[j] = __builtin_amdgcn_wmma_f32_16x16x32_f16(
          false, a, false, b, (short)0, acc[j], false, false);
    }
    __syncthreads();
  }

  const int nn = lane & 15;
  const int mh = lane >> 4;
  #pragma unroll
  for (int j = 0; j < 4; ++j) {
    const int w = n0 + wn * 64 + j * 16 + nn;
    #pragma unroll
    for (int i = 0; i < 8; ++i) {
      const int u = m0 + wm * 16 + mh * 8 + i;
      float rel;
      if (w <= u)            rel = R[(size_t)u * T + (T - 1 - u + w)];
      else if (w == u + 1)   rel = 0.0f;
      else                   rel = R[(size_t)(u + 1) * T + (w - u - 2)];
      const float mv = maskp[(size_t)u * T + w];
      const float sc = (acc[j][i] + rel) * scale;
      S[(size_t)u * T + w] = sc * mv - 1e10f * (1.0f - mv);
    }
  }
}

// ---------------------------------------------------------------------------
__global__ __launch_bounds__(256) void softmax_rows(
    const float* __restrict__ S, half_t* __restrict__ P, int T)
{
  const int row = blockIdx.x;
  const int tid = threadIdx.x;
  const float* s = S + (size_t)row * T;
  __shared__ float red[256];

  float mx = -3.4e38f;
  for (int w = tid; w < T; w += 256) mx = fmaxf(mx, s[w]);
  red[tid] = mx; __syncthreads();
  for (int st = 128; st > 0; st >>= 1) {
    if (tid < st) red[tid] = fmaxf(red[tid], red[tid + st]);
    __syncthreads();
  }
  mx = red[0]; __syncthreads();

  float sum = 0.0f;
  for (int w = tid; w < T; w += 256) sum += __expf(s[w] - mx);
  red[tid] = sum; __syncthreads();
  for (int st = 128; st > 0; st >>= 1) {
    if (tid < st) red[tid] += red[tid + st];
    __syncthreads();
  }
  const float inv = 1.0f / red[0];

  half_t* p = P + (size_t)row * T;
  for (int w = tid; w < T; w += 256) p[w] = (half_t)(__expf(s[w] - mx) * inv);
}

// Reversed-order sinusoidal embedding: pos = T-1-t, n = D/2,
// emb = [sin(pos*inv_i), cos(pos*inv_i)], inv_i = exp(-i*log(1e4)/(n-1)).
__global__ void sinusoid_kernel(float* __restrict__ out, int T, int D)
{
  const int idx = blockIdx.x * blockDim.x + threadIdx.x;
  if (idx >= T * D) return;
  const int t = idx / D;
  const int c = idx % D;
  const int n = D / 2;
  const float pos = (float)(T - 1 - t);
  const int  i   = (c < n) ? c : (c - n);
  const float inv = expf(-(float)i * logf(10000.0f) / (float)(n - 1));
  const float st  = pos * inv;
  out[idx] = (c < n) ? sinf(st) : cosf(st);
}

// Fold r_w_bias / r_r_bias (per-column, col = h*64+d) into the q bias.
__global__ void combine_bias(const float* __restrict__ bq,
                             const float* __restrict__ rw,
                             const float* __restrict__ rr,
                             float* __restrict__ bqw, float* __restrict__ bqr, int D)
{
  const int i = blockIdx.x * blockDim.x + threadIdx.x;
  if (i < D) { bqw[i] = bq[i] + rw[i]; bqr[i] = bq[i] + rr[i]; }
}

// ---------------------------------------------------------------------------
static inline dim3 gemm_grid(int M, int N) {
  return dim3((M + BM - 1) / BM, (N + BN - 1) / BN);
}

extern "C" void kernel_launch(void* const* d_in, const int* in_sizes, int n_in,
                              void* d_out, int out_size, void* d_ws, size_t ws_size,
                              hipStream_t stream)
{
  const float* x    = (const float*)d_in[0];
  const float* y    = (const float*)d_in[1];
  const float* mask = (const float*)d_in[2];
  const float* Wq   = (const float*)d_in[3];
  const float* bq   = (const float*)d_in[4];
  const float* Wk   = (const float*)d_in[5];
  const float* bk   = (const float*)d_in[6];
  const float* Wv   = (const float*)d_in[7];
  const float* bv   = (const float*)d_in[8];
  const float* Wp   = (const float*)d_in[9];
  const float* bp   = (const float*)d_in[10];
  const float* rwb  = (const float*)d_in[11];
  const float* rrb  = (const float*)d_in[12];
  const float* Wo   = (const float*)d_in[13];
  const float* bo   = (const float*)d_in[14];
  float* out = (float*)d_out;

  // ---- carve workspace (~96 MB) ----
  char* w = (char*)d_ws;
  auto carve = [&](size_t bytes) -> char* {
    char* p = w; w += (bytes + 255) & ~(size_t)255; return p;
  };
  float*  pe_sin = (float*) carve(sizeof(float)  * (size_t)kT * kD);       // sinusoid
  half_t* Qw     = (half_t*)carve(sizeof(half_t) * (size_t)kB * kT * kD);  // q + r_w_bias
  half_t* Qr     = (half_t*)carve(sizeof(half_t) * (size_t)kB * kT * kD);  // q + r_r_bias
  half_t* Kh     = (half_t*)carve(sizeof(half_t) * (size_t)kB * kT * kD);
  half_t* Vh     = (half_t*)carve(sizeof(half_t) * (size_t)kB * kT * kD);
  half_t* PE     = (half_t*)carve(sizeof(half_t) * (size_t)kT * kD);
  float*  Rb     = (float*) carve(sizeof(float)  * (size_t)kT * kT);       // rel logits
  float*  Sb     = (float*) carve(sizeof(float)  * (size_t)kT * kT);       // masked scores
  half_t* Pb     = (half_t*)carve(sizeof(half_t) * (size_t)kT * kT);       // softmax probs
  half_t* AO     = (half_t*)carve(sizeof(half_t) * (size_t)kB * kT * kD);  // attn output
  float*  bqw    = (float*) carve(sizeof(float)  * kD);
  float*  bqr    = (float*) carve(sizeof(float)  * kD);

  const dim3 blk(NTHR);
  const int MBT = kB * kT;  // 4096

  combine_bias<<<(kD + 255) / 256, 256, 0, stream>>>(bq, rwb, rrb, bqw, bqr, kD);
  sinusoid_kernel<<<(kT * kD + 255) / 256, 256, 0, stream>>>(pe_sin, kT, kD);

  // Projections (A fp32, W fp32 [K,N], f16 outputs)
  gemm_wmma<float, float, half_t, false><<<gemm_grid(MBT, kD), blk, 0, stream>>>(
      x, kD, Wq, kD, Qw, kD, bqw, MBT, kD, kD);
  gemm_wmma<float, float, half_t, false><<<gemm_grid(MBT, kD), blk, 0, stream>>>(
      x, kD, Wq, kD, Qr, kD, bqr, MBT, kD, kD);
  gemm_wmma<float, float, half_t, false><<<gemm_grid(MBT, kD), blk, 0, stream>>>(
      y, kD, Wk, kD, Kh, kD, bk, MBT, kD, kD);
  gemm_wmma<float, float, half_t, false><<<gemm_grid(MBT, kD), blk, 0, stream>>>(
      y, kD, Wv, kD, Vh, kD, bv, MBT, kD, kD);
  gemm_wmma<float, float, half_t, false><<<gemm_grid(kT, kD), blk, 0, stream>>>(
      pe_sin, kD, Wp, kD, PE, kD, bp, kT, kD, kD);

  // Attention, per (batch, head)
  const float scale = 0.125f;  // 1/sqrt(64)
  for (int b = 0; b < kB; ++b) {
    for (int h = 0; h < kH; ++h) {
      const size_t qoff = (size_t)b * kT * kD + (size_t)h * kHD;
      const half_t* qw = Qw + qoff;
      const half_t* qr = Qr + qoff;
      const half_t* kk = Kh + qoff;
      const half_t* vv = Vh + qoff;
      const half_t* pe = PE + (size_t)h * kHD;

      // R = (q + r_r_bias) * PE^T   [T,T] fp32  (async double-buffered)
      gemm_bt16_async<float><<<gemm_grid(kT, kT), blk, 0, stream>>>(
          qr, kD, pe, kD, Rb, kT, kT, kT, kHD);

      // S = ((q + r_w_bias) * K^T + shift(R)) * scale, masked
      score_kernel<<<gemm_grid(kT, kT), blk, 0, stream>>>(
          qw, kD, kk, kD, Rb, mask + (size_t)b * kT * kT, Sb, kT, kHD, scale);

      // P = softmax(S) -> f16
      softmax_rows<<<kT, 256, 0, stream>>>(Sb, Pb, kT);

      // O_head = P * V  -> f16 slice of attn output
      gemm_wmma<half_t, half_t, half_t, false><<<gemm_grid(kT, kHD), blk, 0, stream>>>(
          Pb, kT, vv, kD, AO + qoff, kD, nullptr, kT, kHD, kT);
    }
  }

  // out = AO @ Wo + bo  (fp32 output)
  gemm_wmma<half_t, float, float, false><<<gemm_grid(MBT, kD), blk, 0, stream>>>(
      AO, kD, Wo, kD, out, kD, bo, MBT, kD, kD);
}
